// LSTMLayer_88587995447887
// MI455X (gfx1250) — matrix operand order repaired and checked by reading, not compiled, not run on hardware
//
#include <hip/hip_runtime.h>
#include <hip/hip_bf16.h>

// LSTM layer: T=1024, B=64, I=H=256, gates=4H=1024.
// Phase 1: x_proj[T*B,4H] = x @ W_ih^T + (b_ih+b_hh)   (WMMA bf16 GEMM, bw-bound)
// Phase 2: persistent single-workgroup scan. Wave owns all 4 gate groups for
//          its (batch-half, h-column) tile -> LSTM pointwise in WMMA
//          accumulator registers. x_proj[t+1] is DMA'd into LDS by the
//          Tensor Data Mover while step t computes (TENSORcnt-tracked).

typedef __bf16 bf16_t;
typedef __attribute__((ext_vector_type(16))) __bf16 v16bf;
typedef __attribute__((ext_vector_type(8)))  __bf16 v8bf;
typedef __attribute__((ext_vector_type(8)))  float  v8f;
typedef __attribute__((ext_vector_type(4)))  unsigned int v4u;
typedef __attribute__((ext_vector_type(8)))  int v8i;
typedef __attribute__((ext_vector_type(4)))  int v4i;

static constexpr int T_  = 1024;
static constexpr int B_  = 64;
static constexpr int I_  = 256;
static constexpr int H_  = 256;
static constexpr int G4  = 1024;   // 4*H

// ---------- fragment loaders (CDNA5 wave32 WMMA VGPR layouts) ----------
// A (16x32 bf16, MxK): lane L holds row M=L%16; K runs [klo,klo+8) and
// [klo+16,klo+24) with klo = (L<16 ? 0 : 8). Two contiguous 16B loads.
__device__ __forceinline__ v16bf load_frag_a(const bf16_t* __restrict__ base,
                                             int ld, int m, int k0, int lane) {
  const int klo = (lane < 16) ? 0 : 8;
  const bf16_t* p = base + (size_t)m * ld + k0 + klo;
  v8bf lo = *(const v8bf*)(p);
  v8bf hi = *(const v8bf*)(p + 16);
  return __builtin_shufflevector(lo, hi, 0,1,2,3,4,5,6,7,8,9,10,11,12,13,14,15);
}

// B (32x16 bf16, KxN): lane L holds column N=L%16; K runs [klo,klo+8) and
// [klo+16,klo+24) with klo = (L<16 ? 0 : 8). B[k,n] = W[n,k] (we multiply by W^T), so
// a column of B is a contiguous row of W. Two contiguous 16B loads.
__device__ __forceinline__ v16bf load_frag_b(const bf16_t* __restrict__ base,
                                             int ld, int n, int k0, int lane) {
  const int koff = (lane < 16) ? 0 : 8;
  const bf16_t* p = base + (size_t)n * ld + k0 + koff;
  v8bf lo = *(const v8bf*)(p);
  v8bf hi = *(const v8bf*)(p + 16);
  return __builtin_shufflevector(lo, hi, 0,1,2,3,4,5,6,7,8,9,10,11,12,13,14,15);
}

__device__ __forceinline__ v8f wmma_bf16(v16bf a, v16bf b, v8f c) {
  return __builtin_amdgcn_wmma_f32_16x16x32_bf16(
      /*neg_a=*/false, a, /*neg_b=*/false, b,
      /*c_mod=*/(short)0, c, /*reuse_a=*/false, /*reuse_b=*/false);
}

__device__ __forceinline__ float sigmoid_f(float x) {
  return 1.0f / (1.0f + __expf(-x));
}
__device__ __forceinline__ float tanh_f(float x) {
  return 1.0f - 2.0f / (1.0f + __expf(2.0f * x));  // stable at both tails
}

// ---- Tensor Data Mover: DMA a contiguous 64x1024 fp32 tile (256 KB) to LDS.
// D# per CDNA5 ISA ch.8: group0 = {flags, lds_addr, global_addr, type=2},
// group1 = dims/strides. 2D tile: 16 rows x 4096 elem, stride 4096, 4B elems.
// Toolchain uses the 6-arg builtin: (v4u, v8i, v4i, v4i, v8i, i32 cpol).
__device__ __forceinline__ void tdm_load_xp(const float* src, unsigned lds_byte) {
  const unsigned long long ga = (unsigned long long)(uintptr_t)src;
  v4u g0;
  g0[0] = 1u;                                   // count=1, user mode, no gather
  g0[1] = lds_byte;                             // lds_addr (bytes)
  g0[2] = (unsigned)(ga & 0xffffffffu);         // global_addr[31:0]
  g0[3] = (unsigned)((ga >> 32) & 0x01ffffffu)  // global_addr[56:32]
          | 0x80000000u;                        // type=2 ("image") in [127:126]
  v8i g1;
  g1[0] = (int)(2u << 16);                      // data_size=4B, no flags
  g1[1] = (int)(4096u << 16);                   // tensor_dim0[15:0]=4096
  g1[2] = (int)(16u << 16);                     // dim0 hi=0 | tensor_dim1=16
  g1[3] = (int)(4096u << 16);                   // dim1 hi=0 | tile_dim0=4096
  g1[4] = 16;                                   // tile_dim1=16, tile_dim2=0
  g1[5] = 4096;                                 // tensor_dim0_stride[31:0]
  g1[6] = 0;                                    // stride0 hi | stride1 lo
  g1[7] = 0;                                    // stride1 hi
  v4i g2 = {0, 0, 0, 0};                        // 2D: groups 2/3 unused
  v4i g3 = {0, 0, 0, 0};
  v8i g4 = {0, 0, 0, 0, 0, 0, 0, 0};            // extra group (6-arg variant)
  __builtin_amdgcn_tensor_load_to_lds(g0, g1, g2, g3, g4, 0);
}

// ---------------------- kernel 0: fp32 -> bf16 convert ----------------------
__global__ void cvt_to_bf16(const float* __restrict__ in,
                            bf16_t* __restrict__ out, int n) {
  int i = blockIdx.x * blockDim.x + threadIdx.x;
  if (i < n) out[i] = (bf16_t)in[i];
}

// -------------- kernel 1: x_proj = x(bf16) @ W_ih^T(bf16) + bias --------------
__global__ __launch_bounds__(256) void xproj_gemm(
    const bf16_t* __restrict__ xbf,     // [T*B, I]
    const bf16_t* __restrict__ wihbf,   // [4H, I]
    const float*  __restrict__ b_ih,    // [4H]
    const float*  __restrict__ b_hh,    // [4H]
    float* __restrict__ xproj)          // [T*B, 4H]
{
  const int lane = threadIdx.x & 31;
  const int wave = threadIdx.x >> 5;
  const int m0 = blockIdx.x * 128 + (wave >> 1) * 32;
  const int n0 = blockIdx.y * 128 + (wave & 1) * 64;

  v8f acc[2][4];
  #pragma unroll
  for (int j = 0; j < 4; ++j) {
    const int col = n0 + j * 16 + (lane & 15);
    const float bias = b_ih[col] + b_hh[col];
    #pragma unroll
    for (int i = 0; i < 2; ++i)
      #pragma unroll
      for (int e = 0; e < 8; ++e)
        acc[i][j][e] = bias;
  }

  #pragma unroll
  for (int k0 = 0; k0 < I_; k0 += 32) {
    v16bf a[2], b[4];
    #pragma unroll
    for (int i = 0; i < 2; ++i)
      a[i] = load_frag_a(xbf, I_, m0 + i * 16 + (lane & 15), k0, lane);
    #pragma unroll
    for (int j = 0; j < 4; ++j)
      b[j] = load_frag_b(wihbf, I_, n0 + j * 16 + (lane & 15), k0, lane);
    #pragma unroll
    for (int i = 0; i < 2; ++i)
      #pragma unroll
      for (int j = 0; j < 4; ++j)
        acc[i][j] = wmma_bf16(a[i], b[j], acc[i][j]);
  }

  const int moff = (lane < 16) ? 0 : 8;
  #pragma unroll
  for (int j = 0; j < 4; ++j) {
    const int col = n0 + j * 16 + (lane & 15);
    #pragma unroll
    for (int i = 0; i < 2; ++i)
      #pragma unroll
      for (int e = 0; e < 8; ++e)
        xproj[(size_t)(m0 + i * 16 + moff + e) * G4 + col] = acc[i][j][e];
  }
}

// ---------------- kernel 2: persistent recurrent scan (1 WGP) ----------------
// 32 waves. Wave w: wr = w&1 -> batch rows [32wr,32wr+32); nc = w>>1 -> h cols
// [16nc,16nc+16). Accumulators seeded from the LDS-resident x_proj tile; TDM
// streams the next tile in the background, published by the step barrier.
__global__ __launch_bounds__(1024) void lstm_scan(
    const bf16_t* __restrict__ whhbf,   // [4H, H] bf16
    const float*  __restrict__ xproj,   // [T, B, 4H]
    const float*  __restrict__ hx,      // [B, H]
    const float*  __restrict__ cx,      // [B, H]
    float* __restrict__ out)            // outputs[T,B,H] ++ h_f[B,H] ++ c_f[B,H]
{
  __shared__ __align__(32) bf16_t hbuf[B_][H_];    //  32 KB
  __shared__ __align__(128) float xp_lds[B_][G4];  // 256 KB (total 288 < 320 KB)

  const int tid  = threadIdx.x;
  const int lane = tid & 31;
  const int wave = tid >> 5;
  const int wr   = wave & 1;
  const int nc   = wave >> 1;
  const int m0   = wr * 32;
  const int moff = (lane < 16) ? 0 : 8;
  const int ncol = nc * 16 + (lane & 15);

  const unsigned xp_lds_byte = (unsigned)(uintptr_t)&xp_lds[0][0];

  // TDM: start DMA of x_proj[0] while we initialize state
  if (wave == 0) tdm_load_xp(xproj, xp_lds_byte);

  #pragma unroll
  for (int r = 0; r < 16; ++r) {
    const int e = tid + 1024 * r;
    hbuf[e >> 8][e & 255] = (bf16_t)hx[e];
  }
  float c_reg[2][8];
  #pragma unroll
  for (int i = 0; i < 2; ++i)
    #pragma unroll
    for (int e = 0; e < 8; ++e)
      c_reg[i][e] = cx[(m0 + i * 16 + moff + e) * H_ + ncol];

  if (wave == 0) __builtin_amdgcn_s_wait_tensorcnt(0);
  __syncthreads();

  float* __restrict__ hfinal = out + (size_t)T_ * B_ * H_;
  float* __restrict__ cfinal = hfinal + B_ * H_;

  for (int t = 0; t < T_; ++t) {
    // ---- seed accumulators with x_proj[t] from LDS ----
    v8f acc[2][4];
    #pragma unroll
    for (int i = 0; i < 2; ++i)
      #pragma unroll
      for (int g = 0; g < 4; ++g)
        #pragma unroll
        for (int e = 0; e < 8; ++e)
          acc[i][g][e] = xp_lds[m0 + i * 16 + moff + e][g * 256 + ncol];

    // ---- gates += h @ W_hh^T ----
    #pragma unroll
    for (int k0 = 0; k0 < H_; k0 += 32) {
      v16bf a[2], b[4];
      #pragma unroll
      for (int i = 0; i < 2; ++i)
        a[i] = load_frag_a(&hbuf[0][0], H_, m0 + i * 16 + (lane & 15), k0, lane);
      #pragma unroll
      for (int g = 0; g < 4; ++g)
        b[g] = load_frag_b(whhbf, H_, g * 256 + nc * 16 + (lane & 15), k0, lane);
      #pragma unroll
      for (int i = 0; i < 2; ++i)
        #pragma unroll
        for (int g = 0; g < 4; ++g)
          acc[i][g] = wmma_bf16(a[i], b[g], acc[i][g]);
    }
    __syncthreads();   // all waves done reading xp_lds(t) and hbuf(t)

    // ---- TDM: overlap next tile's DMA with the pointwise phase ----
    if (wave == 0 && t + 1 < T_)
      tdm_load_xp(xproj + (size_t)(t + 1) * (B_ * G4), xp_lds_byte);

    // ---- fused LSTM pointwise in registers (gate order i,f,g,o) ----
    float* __restrict__ outp = out + (size_t)t * (B_ * H_);
    #pragma unroll
    for (int i = 0; i < 2; ++i) {
      #pragma unroll
      for (int e = 0; e < 8; ++e) {
        const int row = m0 + i * 16 + moff + e;   // batch index
        const float ig = sigmoid_f(acc[i][0][e]);
        const float fg = sigmoid_f(acc[i][1][e]);
        const float gg = tanh_f  (acc[i][2][e]);
        const float og = sigmoid_f(acc[i][3][e]);
        const float cn = fg * c_reg[i][e] + ig * gg;
        const float hn = og * tanh_f(cn);
        c_reg[i][e] = cn;
        hbuf[row][ncol] = (bf16_t)hn;
        outp[row * H_ + ncol] = hn;
        if (t == T_ - 1) hfinal[row * H_ + ncol] = hn;
      }
    }
    if (wave == 0) __builtin_amdgcn_s_wait_tensorcnt(0);
    __syncthreads();   // publishes hbuf(t+1) and the DMA'd xp_lds(t+1)
  }

  #pragma unroll
  for (int i = 0; i < 2; ++i)
    #pragma unroll
    for (int e = 0; e < 8; ++e)
      cfinal[(m0 + i * 16 + moff + e) * H_ + ncol] = c_reg[i][e];
}

// ------------------------------- launch glue -------------------------------
extern "C" void kernel_launch(void* const* d_in, const int* in_sizes, int n_in,
                              void* d_out, int out_size, void* d_ws, size_t ws_size,
                              hipStream_t stream) {
  const float* x    = (const float*)d_in[0];   // [T,B,I]
  const float* hx   = (const float*)d_in[1];   // [B,H]
  const float* cx   = (const float*)d_in[2];   // [B,H]
  const float* W_ih = (const float*)d_in[3];   // [4H,I]
  const float* W_hh = (const float*)d_in[4];   // [4H,H]
  const float* b_ih = (const float*)d_in[5];   // [4H]
  const float* b_hh = (const float*)d_in[6];   // [4H]
  float* out = (float*)d_out;

  const size_t nX = (size_t)T_ * B_ * I_;      // 16,777,216
  const size_t nW = (size_t)G4 * I_;           // 262,144
  char* ws = (char*)d_ws;
  bf16_t* xbf   = (bf16_t*)(ws);
  bf16_t* wihbf = (bf16_t*)(ws + nX * 2);
  bf16_t* whhbf = (bf16_t*)(ws + nX * 2 + nW * 2);
  float*  xproj = (float*) (ws + nX * 2 + nW * 4);

  cvt_to_bf16<<<(int)(nX / 256), 256, 0, stream>>>(x,    xbf,   (int)nX);
  cvt_to_bf16<<<(int)(nW / 256), 256, 0, stream>>>(W_ih, wihbf, (int)nW);
  cvt_to_bf16<<<(int)(nW / 256), 256, 0, stream>>>(W_hh, whhbf, (int)nW);

  dim3 g1((T_ * B_) / 128, G4 / 128);          // 512 x 8 blocks
  xproj_gemm<<<g1, 256, 0, stream>>>(xbf, wihbf, b_ih, b_hh, xproj);

  lstm_scan<<<1, 1024, 0, stream>>>(whhbf, xproj, hx, cx, out);
}
